// GeneralistModel_50251117363212
// MI455X (gfx1250) — matrix operand order repaired and checked by reading, not compiled, not run on hardware
//
#include <hip/hip_runtime.h>
#include <hip/hip_bf16.h>

// ---------------------------------------------------------------------------
// Types for CDNA5 WMMA (wave32): V_WMMA_F32_16X16X32_BF16
// ---------------------------------------------------------------------------
typedef __bf16 bf16_t;
typedef __attribute__((ext_vector_type(16))) __bf16 v16bf;
typedef __attribute__((ext_vector_type(8)))  __bf16 v8bf;
typedef __attribute__((ext_vector_type(4)))  __bf16 v4bf;
typedef __attribute__((ext_vector_type(8)))  float  v8f;
typedef __attribute__((ext_vector_type(4)))  float  v4f;

union Frag16 { v16bf v; v8bf h[2]; };

// Problem constants
#define BB 256
#define LL 256
#define SS 253
#define SP 256   // padded S
#define DD 768
#define HH 128

// ---------------------------------------------------------------------------
// K1: transpose + convert weight f32 [K,N] -> bf16 [N,K]
// grid (DD/256, DD), block 256
// ---------------------------------------------------------------------------
__global__ __launch_bounds__(256)
void wconv_kernel(const float* __restrict__ W, bf16_t* __restrict__ WT) {
    int k = blockIdx.x * 256 + threadIdx.x;   // 0..767
    int n = blockIdx.y;                        // 0..767
    WT[(size_t)n * DD + k] = (bf16_t)W[(size_t)k * DD + n];
}

// ---------------------------------------------------------------------------
// K2: convert f32 -> bf16, 4 elements/thread (n must be multiple of 4)
// ---------------------------------------------------------------------------
__global__ __launch_bounds__(256)
void f2bf_kernel(const float* __restrict__ src, bf16_t* __restrict__ dst, long n4) {
    long i = (long)blockIdx.x * 256 + threadIdx.x;
    if (i >= n4) return;
    v4f f = *(const v4f*)(src + i * 4);
    v4bf o;
    o.x = (bf16_t)f.x; o.y = (bf16_t)f.y; o.z = (bf16_t)f.z; o.w = (bf16_t)f.w;
    *(v4bf*)(dst + i * 4) = o;
}

// ---------------------------------------------------------------------------
// K3: convert candidate embeds to padded bf16 [B,256,768] and compute
//     vm[b,s] = cand_mask ? dot(cand[b,s,:], WV) : 0  (pad rows -> 0)
// grid (SP, B), block 256
// ---------------------------------------------------------------------------
__global__ __launch_bounds__(256)
void candconv_kernel(const float* __restrict__ cand, const int* __restrict__ cmask,
                     const float* __restrict__ WV, bf16_t* __restrict__ candbf,
                     float* __restrict__ vm) {
    __shared__ float red[256];
    int s = blockIdx.x, b = blockIdx.y, t = threadIdx.x;
    float partial = 0.0f;
    bf16_t* drow = candbf + ((size_t)b * SP + s) * DD;
    if (s < SS) {
        const float* srow = cand + ((size_t)b * SS + s) * DD;
        #pragma unroll
        for (int d = t; d < DD; d += 256) {
            float val = srow[d];
            drow[d] = (bf16_t)val;
            partial += val * WV[d];
        }
    } else {
        for (int d = t; d < DD; d += 256) drow[d] = (bf16_t)0.0f;
    }
    red[t] = partial;
    __syncthreads();
    for (int off = 128; off > 0; off >>= 1) {
        if (t < off) red[t] += red[t + off];
        __syncthreads();
    }
    if (t == 0) {
        float v = 0.0f;
        if (s < SS && cmask[b * SS + s] != 0) v = red[0];
        vm[(size_t)b * SP + s] = v;
    }
}

// ---------------------------------------------------------------------------
// K4: bf16 GEMM  C[M,N] = A[M,K] * BT[N,K]^T, f32 accumulate, bf16 out.
// Block = 256 threads = 8 waves; wave w -> rows m0..m0+15, cols n0..n0+63.
// grid.x = M/128, grid.y = N/64.  K multiple of 32.
// ---------------------------------------------------------------------------
__global__ __launch_bounds__(256)
void gemm_bf16_kernel(const bf16_t* __restrict__ A, const bf16_t* __restrict__ BT,
                      bf16_t* __restrict__ C, int M, int N, int K) {
    const int lane = threadIdx.x & 31;
    const int wave = threadIdx.x >> 5;
    const int m0 = (blockIdx.x * 8 + wave) * 16;
    const int n0 = blockIdx.y * 64;

    // A fragment addressing (16x32 bf16, ISA layout):
    // lanes 0-15: M=lane, K chunks {0..7,16..23}; lanes 16-31: M=lane-16, {8..15,24..31}
    const int am    = m0 + (lane & 15);
    const int aksel = (lane >> 4) * 8;
    const bf16_t* arow = A + (size_t)am * K + aksel;

    v8f acc[4] = {v8f{}, v8f{}, v8f{}, v8f{}};

    for (int k0 = 0; k0 < K; k0 += 32) {
        Frag16 a;
        a.h[0] = *(const v8bf*)(arow + k0);
        a.h[1] = *(const v8bf*)(arow + k0 + 16);
        #pragma unroll
        for (int t = 0; t < 4; ++t) {
            // B fragment (32x16): lanes 0-15: N=lane, k={0..7,16..23}; lanes 16-31: N=lane-16, k={8..15,24..31}
            const bf16_t* bp = BT + (size_t)(n0 + t * 16 + (lane & 15)) * K
                                  + k0 + (lane >> 4) * 8;
            Frag16 bfr;
            bfr.h[0] = *(const v8bf*)(bp);
            bfr.h[1] = *(const v8bf*)(bp + 16);
            acc[t] = __builtin_amdgcn_wmma_f32_16x16x32_bf16(
                false, a.v, false, bfr.v, (short)0, acc[t], false, false);
        }
    }

    // C layout: lanes 0-15: N=lane, M=r; lanes 16-31: N=lane-16, M=r+8
    const int cn = lane & 15;
    const int mh = (lane >> 4) * 8;
    #pragma unroll
    for (int t = 0; t < 4; ++t)
        #pragma unroll
        for (int r = 0; r < 8; ++r)
            C[(size_t)(m0 + mh + r) * N + n0 + t * 16 + cn] = (bf16_t)acc[t][r];
}

// ---------------------------------------------------------------------------
// K5: attention:  x[b,l] = sum_s sigmoid( (Q[b,l]·K[b,s]) * scale ) * vm[b,s]
// Qbf rows = b*256+l, Kbf rows = b*256+s (bf16, padded; vm=0 on pads).
// grid (B, L/128), block 256 = 8 waves; wave -> 16 rows of l, loops all s.
// ---------------------------------------------------------------------------
__global__ __launch_bounds__(256)
void attn_kernel(const bf16_t* __restrict__ Qbf, const bf16_t* __restrict__ Kbf,
                 const float* __restrict__ vm, float* __restrict__ x) {
    const int lane = threadIdx.x & 31;
    const int wave = threadIdx.x >> 5;
    const int b  = blockIdx.x;
    const int l0 = blockIdx.y * 128 + wave * 16;
    const float scale = 0.03608439182435161f;  // 1/sqrt(768)

    const int am = b * SP + l0 + (lane & 15);
    const bf16_t* arow = Qbf + (size_t)am * DD + (lane >> 4) * 8;

    float xacc[8] = {0, 0, 0, 0, 0, 0, 0, 0};

    for (int n0 = 0; n0 < SP; n0 += 16) {
        const bf16_t* brow = Kbf + (size_t)(b * SP + n0 + (lane & 15)) * DD
                                 + (lane >> 4) * 8;
        v8f acc = {};
        for (int k0 = 0; k0 < DD; k0 += 32) {
            Frag16 a, bfr;
            a.h[0]   = *(const v8bf*)(arow + k0);
            a.h[1]   = *(const v8bf*)(arow + k0 + 16);
            bfr.h[0] = *(const v8bf*)(brow + k0);
            bfr.h[1] = *(const v8bf*)(brow + k0 + 16);
            acc = __builtin_amdgcn_wmma_f32_16x16x32_bf16(
                false, a.v, false, bfr.v, (short)0, acc, false, false);
        }
        float vv = vm[(size_t)b * SP + n0 + (lane & 15)];
        #pragma unroll
        for (int r = 0; r < 8; ++r) {
            float p = 1.0f / (1.0f + __expf(-acc[r] * scale));
            xacc[r] += p * vv;
        }
    }

    // Reduce the partial sums over the 16 lanes of each half-group (wave32 shuffles)
    #pragma unroll
    for (int r = 0; r < 8; ++r) {
        float v = xacc[r];
        v += __shfl_xor(v, 1, 16);
        v += __shfl_xor(v, 2, 16);
        v += __shfl_xor(v, 4, 16);
        v += __shfl_xor(v, 8, 16);
        if ((lane & 15) == 0)
            x[(size_t)b * LL + l0 + (lane >> 4) * 8 + r] = v;
    }
}

// ---------------------------------------------------------------------------
// K6: head: h = (ctx_mask * x) @ W1 + b1 ; y = h @ W2 + b2
// grid B, block 128
// ---------------------------------------------------------------------------
__global__ __launch_bounds__(128)
void head_kernel(const float* __restrict__ x, const int* __restrict__ ctx_mask,
                 const float* __restrict__ W1, const float* __restrict__ b1,
                 const float* __restrict__ W2, const float* __restrict__ b2,
                 float* __restrict__ out) {
    __shared__ float hbuf[HH];
    int b = blockIdx.x, h = threadIdx.x;
    float acc = b1[h];
    for (int l = 0; l < LL; ++l) {
        float xm = (ctx_mask[b * LL + l] != 0) ? x[(size_t)b * LL + l] : 0.0f;
        acc += xm * W1[l * HH + h];
    }
    hbuf[h] = acc;
    __syncthreads();
    if (h < 5) {
        float y = b2[h];
        for (int j = 0; j < HH; ++j) y += hbuf[j] * W2[j * 5 + h];
        out[b * 5 + h] = y;
    }
}

// ---------------------------------------------------------------------------
// Launch
// ---------------------------------------------------------------------------
extern "C" void kernel_launch(void* const* d_in, const int* in_sizes, int n_in,
                              void* d_out, int out_size, void* d_ws, size_t ws_size,
                              hipStream_t stream) {
    (void)in_sizes; (void)n_in; (void)out_size; (void)ws_size;
    const float* ctx       = (const float*)d_in[0];   // [B,L,D]
    const float* cand      = (const float*)d_in[1];   // [B,S,D]
    const int*   ctx_mask  = (const int*)  d_in[2];   // [B,L]
    const int*   cand_mask = (const int*)  d_in[3];   // [B,S]
    const float* WK        = (const float*)d_in[4];   // [D,D]
    const float* WQ        = (const float*)d_in[5];   // [D,D]
    const float* WV        = (const float*)d_in[6];   // [D,1]
    const float* W1        = (const float*)d_in[7];   // [L,H]
    const float* b1        = (const float*)d_in[8];   // [H]
    const float* W2        = (const float*)d_in[9];   // [H,5]
    const float* b2        = (const float*)d_in[10];  // [5]
    float* out = (float*)d_out;

    // Workspace layout (offsets 256B aligned); bufA reused (ctxbf then candbf)
    char* ws = (char*)d_ws;
    size_t off = 0;
    const size_t wsz   = (size_t)DD * DD * sizeof(bf16_t);      // 1.125 MB
    const size_t bigsz = (size_t)BB * SP * DD * sizeof(bf16_t); // 96 MB
    bf16_t* WQt  = (bf16_t*)(ws + off); off += wsz;
    bf16_t* WKt  = (bf16_t*)(ws + off); off += wsz;
    float*  vm   = (float*)(ws + off);  off += (size_t)BB * SP * sizeof(float);
    float*  xbuf = (float*)(ws + off);  off += (size_t)BB * LL * sizeof(float);
    bf16_t* bufA = (bf16_t*)(ws + off); off += bigsz;  // ctxbf -> candbf
    bf16_t* bufB = (bf16_t*)(ws + off); off += bigsz;  // Qbf
    bf16_t* bufC = (bf16_t*)(ws + off); off += bigsz;  // Kbf

    // 1) weights -> transposed bf16
    wconv_kernel<<<dim3(DD / 256, DD), 256, 0, stream>>>(WQ, WQt);
    wconv_kernel<<<dim3(DD / 256, DD), 256, 0, stream>>>(WK, WKt);

    // 2) ctx -> bf16
    {
        long n4 = (long)BB * LL * DD / 4;
        f2bf_kernel<<<(unsigned)((n4 + 255) / 256), 256, 0, stream>>>(ctx, bufA, n4);
    }

    // 3) Q = ctxbf @ WQ   (M = B*L = 65536)
    gemm_bf16_kernel<<<dim3(BB * LL / 128, DD / 64), 256, 0, stream>>>(
        bufA, WQt, bufB, BB * LL, DD, DD);

    // 4) cand -> padded bf16 (+ vm)   [bufA safe to overwrite: same-stream order]
    candconv_kernel<<<dim3(SP, BB), 256, 0, stream>>>(cand, cand_mask, WV, bufA, vm);

    // 5) K = candbf @ WK  (M = B*Spad = 65536)
    gemm_bf16_kernel<<<dim3(BB * SP / 128, DD / 64), 256, 0, stream>>>(
        bufA, WKt, bufC, BB * SP, DD, DD);

    // 6) attention -> x
    attn_kernel<<<dim3(BB, LL / 128), 256, 0, stream>>>(bufB, bufC, vm, xbuf);

    // 7) head -> out
    head_kernel<<<BB, 128, 0, stream>>>(xbuf, ctx_mask, W1, b1, W2, b2, out);
}
